// WanSelfAttention_80530636800312
// MI455X (gfx1250) — compile-verified, matrix operand order checked by reading, not run on hardware
//
#include <hip/hip_runtime.h>
#include <hip/hip_bf16.h>

typedef __bf16 bf16_t;
typedef __attribute__((ext_vector_type(8)))  bf16_t v8bf;
typedef __attribute__((ext_vector_type(16))) bf16_t v16bf;
typedef __attribute__((ext_vector_type(8)))  float  v8f;

#define S_LEN 3072
#define DIMSZ 2048
#define NHEAD 16
#define HDIM  128
#define BK    32

#if __has_builtin(__builtin_amdgcn_global_load_async_to_lds_b128)
#define HAVE_ASYNC_LDS 1
#endif

#if __has_builtin(__builtin_amdgcn_s_wait_asynccnt)
#define WAIT_ASYNC(n) __builtin_amdgcn_s_wait_asynccnt(n)
#else
#define WAIT_ASYNC(n) asm volatile("s_wait_asynccnt %0" ::"i"(n) : "memory")
#endif

// Builtin expects (v4i32 AS1*, v4i32 AS3*, imm offset, imm cpol)
typedef int v4i __attribute__((vector_size(16)));
typedef __attribute__((address_space(1))) v4i gas_v4i;
typedef __attribute__((address_space(3))) v4i las_v4i;

// 16-byte global -> LDS copy; async (ASYNCcnt-tracked) when available.
static __device__ __forceinline__ void cp16(bf16_t* lds, const bf16_t* g) {
#if defined(HAVE_ASYNC_LDS)
  bf16_t* gm = const_cast<bf16_t*>(g);
  __builtin_amdgcn_global_load_async_to_lds_b128((gas_v4i*)gm, (las_v4i*)lds, 0, 0);
#else
  *reinterpret_cast<v8bf*>(lds) = *reinterpret_cast<const v8bf*>(g);
#endif
}

static __device__ __forceinline__ v16bf load_frag16(const bf16_t* base) {
  // ISA 16-bit A/B fragment: lane<16 -> K{0..7,16..23}, lane>=16 -> K{8..15,24..31}
  union { v16bf v; v8bf h[2]; } u;
  u.h[0] = *reinterpret_cast<const v8bf*>(base);
  u.h[1] = *reinterpret_cast<const v8bf*>(base + 16);
  return u.v;
}

static __device__ __forceinline__ v8f vzero8() {
  v8f v;
  #pragma unroll
  for (int i = 0; i < 8; ++i) v[i] = 0.0f;
  return v;
}

// ---------------------------------------------------------------------------
// f32 -> bf16 elementwise convert
// ---------------------------------------------------------------------------
__global__ void cvt_f32_bf16(const float* __restrict__ in, bf16_t* __restrict__ out, int n) {
  int i = blockIdx.x * blockDim.x + threadIdx.x;
  if (i < n) out[i] = (bf16_t)in[i];
}

// ---------------------------------------------------------------------------
// C[M,N] = A[M,K]*B[N,K]^T + bias[N]  (bf16 in, f32 out)
// Block 256 thr = 8 waves; block tile 128x256; wave tile 64x64 (4x4 WMMA).
// A/B tiles staged to LDS via async-to-LDS, double buffered.
// ---------------------------------------------------------------------------
__global__ __launch_bounds__(256) void gemm_bf16_nt(
    const bf16_t* __restrict__ A, const bf16_t* __restrict__ B,
    const float* __restrict__ bias, float* __restrict__ C,
    int M, int Ncols, int K)
{
  __shared__ __align__(16) bf16_t lA[2][128][BK];
  __shared__ __align__(16) bf16_t lB[2][256][BK];

  const int t     = threadIdx.x;
  const int lane  = t & 31;
  const int w     = t >> 5;             // 0..7
  const int waveM = w >> 2;             // 0..1
  const int waveN = w & 3;              // 0..3
  const int blkM  = blockIdx.y * 128;
  const int blkN  = blockIdx.x * 256;
  const int lrow  = lane & 15;
  const int lko   = (lane < 16) ? 0 : 8;
  const int rhalf = (lane < 16) ? 0 : 8;

  v8f acc[4][4];
  #pragma unroll
  for (int i = 0; i < 4; ++i)
    #pragma unroll
    for (int j = 0; j < 4; ++j) acc[i][j] = vzero8();

  // Per-stage staging: A 512 x16B chunks (2/thread), B 1024 chunks (4/thread)
  auto issue = [&](int buf, int k0) {
    #pragma unroll
    for (int c = 0; c < 2; ++c) {
      const int id = t * 2 + c;
      const int r = id >> 2, co = (id & 3) * 8;
      cp16(&lA[buf][r][co], A + (size_t)(blkM + r) * K + k0 + co);
    }
    #pragma unroll
    for (int c = 0; c < 4; ++c) {
      const int id = t * 4 + c;
      const int r = id >> 2, co = (id & 3) * 8;
      cp16(&lB[buf][r][co], B + (size_t)(blkN + r) * K + k0 + co);
    }
  };

  const int nst = K / BK;
  issue(0, 0);
  for (int i = 0; i < nst; ++i) {
    const int buf = i & 1;
    if (i + 1 < nst) {
      issue(1 - buf, (i + 1) * BK);  // prefetch next stage into other buffer
      WAIT_ASYNC(6);                 // 6 outstanding = next stage only -> this stage landed
    } else {
      WAIT_ASYNC(0);
    }
    __syncthreads();

    v16bf af[4], bfr[4];
    #pragma unroll
    for (int ms = 0; ms < 4; ++ms)
      af[ms] = load_frag16(&lA[buf][waveM * 64 + ms * 16 + lrow][lko]);
    #pragma unroll
    for (int ns = 0; ns < 4; ++ns)
      bfr[ns] = load_frag16(&lB[buf][waveN * 64 + ns * 16 + lrow][lko]);
    #pragma unroll
    for (int ms = 0; ms < 4; ++ms)
      #pragma unroll
      for (int ns = 0; ns < 4; ++ns)
        acc[ms][ns] = __builtin_amdgcn_wmma_f32_16x16x32_bf16(
            false, af[ms], false, bfr[ns], (short)0, acc[ms][ns], false, false);

    __syncthreads();  // all waves done reading buf before it is refilled
  }

  #pragma unroll
  for (int ms = 0; ms < 4; ++ms)
    #pragma unroll
    for (int ns = 0; ns < 4; ++ns) {
      const int col = blkN + waveN * 64 + ns * 16 + lrow;
      const float bv = bias[col];
      #pragma unroll
      for (int e = 0; e < 8; ++e) {
        const int row = blkM + waveM * 64 + ms * 16 + rhalf + e;
        C[(size_t)row * Ncols + col] = acc[ms][ns][e] + bv;
      }
    }
}

// ---------------------------------------------------------------------------
// RMSNorm(row over DIM) + interleaved RoPE + bf16 pack to [N][S][D]
// ---------------------------------------------------------------------------
__global__ __launch_bounds__(256) void norm_rope_pack(
    const float* __restrict__ raw,     // [S][DIM]
    const float* __restrict__ normw,   // [DIM]
    const float* __restrict__ cosb,    // [S][HDIM]
    const float* __restrict__ sinb,    // [S][HDIM]
    bf16_t* __restrict__ dst)          // [NHEAD][S][HDIM]
{
  __shared__ float red[256];
  const int s = blockIdx.x;
  const int t = threadIdx.x;
  const int base = t * 8;
  const float* row = raw + (size_t)s * DIMSZ;

  float x[8];
  float ss = 0.0f;
  #pragma unroll
  for (int j = 0; j < 8; ++j) { x[j] = row[base + j]; ss += x[j] * x[j]; }
  red[t] = ss;
  __syncthreads();
  for (int off = 128; off > 0; off >>= 1) {
    if (t < off) red[t] += red[t + off];
    __syncthreads();
  }
  const float r = rsqrtf(red[0] * (1.0f / (float)DIMSZ) + 1e-6f);

  const int n  = base / HDIM;
  const int d0 = base % HDIM;
  union { v8bf v; bf16_t a[8]; } ou;
  #pragma unroll
  for (int j = 0; j < 8; j += 2) {
    const float xe = x[j]     * r * normw[base + j];
    const float xo = x[j + 1] * r * normw[base + j + 1];
    const float c0 = cosb[s * HDIM + d0 + j];
    const float s0 = sinb[s * HDIM + d0 + j];
    const float c1 = cosb[s * HDIM + d0 + j + 1];
    const float s1 = sinb[s * HDIM + d0 + j + 1];
    ou.a[j]     = (bf16_t)(xe * c0 - xo * s0);
    ou.a[j + 1] = (bf16_t)(xo * c1 + xe * s1);
  }
  bf16_t* dp = dst + ((size_t)n * S_LEN + s) * HDIM + d0;
  *reinterpret_cast<v8bf*>(dp) = ou.v;
}

// ---------------------------------------------------------------------------
// V pack: [S][DIM] f32 -> [N][D][S] bf16 (key index contiguous)
// ---------------------------------------------------------------------------
__global__ __launch_bounds__(256) void pack_v(
    const float* __restrict__ raw, bf16_t* __restrict__ vt)
{
  const int s = blockIdx.x;
  const int t = threadIdx.x;
  const int base = t * 8;
  const int n  = base / HDIM;
  const int d0 = base % HDIM;
  const float* row = raw + (size_t)s * DIMSZ + base;
  #pragma unroll
  for (int j = 0; j < 8; ++j)
    vt[((size_t)n * HDIM + d0 + j) * S_LEN + s] = (bf16_t)row[j];
}

// ---------------------------------------------------------------------------
// Flash attention: head = blockIdx.y, 64 query rows/block (4 waves x 16).
// K/V key-chunks (32) staged to LDS (shared by all 4 waves), double buffered
// async; online softmax; P transposed C-layout->A-layout via per-wave LDS.
// ---------------------------------------------------------------------------
__global__ __launch_bounds__(128) void attn_fwd(
    const bf16_t* __restrict__ Qh,   // [N][S][D]
    const bf16_t* __restrict__ Kh,   // [N][S][D]
    const bf16_t* __restrict__ Vt,   // [N][D][S]
    bf16_t* __restrict__ Abuf,       // [S][DIM]
    const int* __restrict__ seq_lens)
{
  __shared__ __align__(16) bf16_t lK[2][32][HDIM];   // keys x d
  __shared__ __align__(16) bf16_t lV[2][HDIM][32];   // d x keys
  __shared__ __align__(16) bf16_t plds[4][16 * 32];  // per-wave P scratch

  const int t     = threadIdx.x;
  const int lane  = t & 31;
  const int w     = t >> 5;
  const int head  = blockIdx.y;
  const int m0    = blockIdx.x * 64 + w * 16;
  const int lrow  = lane & 15;
  const int lko   = (lane < 16) ? 0 : 8;
  const int rhalf = (lane < 16) ? 0 : 8;
  const int slen  = seq_lens[0];
  const float scale = 0.08838834764831845f;  // 1/sqrt(128)

  // Q fragments (16 rows x 128), 4 k-chunks of 32
  v16bf qf[4];
  const bf16_t* qbase = Qh + ((size_t)head * S_LEN + (m0 + lrow)) * HDIM;
  #pragma unroll
  for (int c = 0; c < 4; ++c) qf[c] = load_frag16(qbase + c * 32 + lko);

  v8f acc[8];
  #pragma unroll
  for (int i = 0; i < 8; ++i) acc[i] = vzero8();
  float mi[8], li[8];
  #pragma unroll
  for (int e = 0; e < 8; ++e) { mi[e] = -3.0e38f; li[e] = 0.0f; }

  // Stage K tile (32x128) + V tile (128x32): 512+512 chunks / 128 thr = 8 each
  auto issueKV = [&](int buf, int kb) {
    #pragma unroll
    for (int c = 0; c < 4; ++c) {
      const int id = t * 4 + c;
      const int r = id >> 4, co = (id & 15) * 8;
      cp16(&lK[buf][r][co], Kh + ((size_t)head * S_LEN + kb + r) * HDIM + co);
    }
    #pragma unroll
    for (int c = 0; c < 4; ++c) {
      const int id = t * 4 + c;
      const int r = id >> 2, co = (id & 3) * 8;
      cp16(&lV[buf][r][co], Vt + ((size_t)head * HDIM + r) * S_LEN + kb + co);
    }
  };

  const int nst = S_LEN / 32;
  issueKV(0, 0);
  for (int kc = 0; kc < nst; ++kc) {
    const int buf = kc & 1;
    const int kb = kc * 32;
    if (kc + 1 < nst) {
      issueKV(1 - buf, kb + 32);
      WAIT_ASYNC(8);
    } else {
      WAIT_ASYNC(0);
    }
    __syncthreads();

    // ---- scores: Q (16x128) x K_chunk^T -> two 16x16 f32 frags
    v8f sc0 = vzero8(), sc1 = vzero8();
    #pragma unroll
    for (int c = 0; c < 4; ++c) {
      v16bf kf0 = load_frag16(&lK[buf][lrow][c * 32 + lko]);
      v16bf kf1 = load_frag16(&lK[buf][16 + lrow][c * 32 + lko]);
      sc0 = __builtin_amdgcn_wmma_f32_16x16x32_bf16(false, qf[c], false, kf0, (short)0, sc0, false, false);
      sc1 = __builtin_amdgcn_wmma_f32_16x16x32_bf16(false, qf[c], false, kf1, (short)0, sc1, false, false);
    }
    // ---- scale + key mask (column == key index)
    const bool ok0 = (kb + lrow) < slen;
    const bool ok1 = (kb + 16 + lrow) < slen;
    #pragma unroll
    for (int e = 0; e < 8; ++e) {
      sc0[e] = ok0 ? sc0[e] * scale : -3.0e38f;
      sc1[e] = ok1 ? sc1[e] * scale : -3.0e38f;
    }
    // ---- online softmax (row stats via width-16 shuffles)
    float cm[8], rs[8], alpha[8];
    #pragma unroll
    for (int e = 0; e < 8; ++e) cm[e] = fmaxf(sc0[e], sc1[e]);
    #pragma unroll
    for (int off = 1; off < 16; off <<= 1)
      #pragma unroll
      for (int e = 0; e < 8; ++e) cm[e] = fmaxf(cm[e], __shfl_xor(cm[e], off, 32));
    #pragma unroll
    for (int e = 0; e < 8; ++e) {
      const float nm = fmaxf(mi[e], cm[e]);
      alpha[e] = __expf(mi[e] - nm);
      mi[e] = nm;
    }
    #pragma unroll
    for (int e = 0; e < 8; ++e) {
      sc0[e] = __expf(sc0[e] - mi[e]);
      sc1[e] = __expf(sc1[e] - mi[e]);
      rs[e]  = sc0[e] + sc1[e];
    }
    #pragma unroll
    for (int off = 1; off < 16; off <<= 1)
      #pragma unroll
      for (int e = 0; e < 8; ++e) rs[e] += __shfl_xor(rs[e], off, 32);
    #pragma unroll
    for (int e = 0; e < 8; ++e) li[e] = li[e] * alpha[e] + rs[e];
    #pragma unroll
    for (int i = 0; i < 8; ++i)
      #pragma unroll
      for (int e = 0; e < 8; ++e) acc[i][e] *= alpha[e];

    // ---- P: C-layout -> row-major 16x32 bf16 (per-wave LDS scratch)
    bf16_t* pl = plds[w];
    #pragma unroll
    for (int e = 0; e < 8; ++e) {
      pl[(rhalf + e) * 32 + lrow]      = (bf16_t)sc0[e];
      pl[(rhalf + e) * 32 + 16 + lrow] = (bf16_t)sc1[e];
    }
    asm volatile("s_wait_dscnt 0x0" ::: "memory");  // same-wave LDS RAW
    union { v16bf v; v8bf h[2]; } pu;
    pu.h[0] = *reinterpret_cast<const v8bf*>(pl + lrow * 32 + lko);
    pu.h[1] = *reinterpret_cast<const v8bf*>(pl + lrow * 32 + lko + 16);
    const v16bf pf = pu.v;

    // ---- acc += P (16x32) x V_chunk (32x128)
    #pragma unroll
    for (int dg = 0; dg < 8; ++dg) {
      v16bf vf = load_frag16(&lV[buf][dg * 16 + lrow][lko]);
      acc[dg] = __builtin_amdgcn_wmma_f32_16x16x32_bf16(false, pf, false, vf, (short)0, acc[dg], false, false);
    }
    __syncthreads();  // everyone done with buf before refill
  }

  // ---- epilogue: divide by l, store bf16 into [S][DIM] token-major
  float inv[8];
  #pragma unroll
  for (int e = 0; e < 8; ++e) inv[e] = 1.0f / li[e];
  #pragma unroll
  for (int dg = 0; dg < 8; ++dg)
    #pragma unroll
    for (int e = 0; e < 8; ++e) {
      const int token = m0 + rhalf + e;
      Abuf[(size_t)token * DIMSZ + head * HDIM + dg * 16 + lrow] =
          (bf16_t)(acc[dg][e] * inv[e]);
    }
}

// ---------------------------------------------------------------------------
extern "C" void kernel_launch(void* const* d_in, const int* in_sizes, int n_in,
                              void* d_out, int out_size, void* d_ws, size_t ws_size,
                              hipStream_t stream) {
  const float* x    = (const float*)d_in[0];
  const float* Wq   = (const float*)d_in[1];
  const float* bq   = (const float*)d_in[2];
  const float* Wk   = (const float*)d_in[3];
  const float* bk   = (const float*)d_in[4];
  const float* Wv   = (const float*)d_in[5];
  const float* bv   = (const float*)d_in[6];
  const float* Wo   = (const float*)d_in[7];
  const float* bo   = (const float*)d_in[8];
  const float* nqw  = (const float*)d_in[9];
  const float* nkw  = (const float*)d_in[10];
  const float* rcos = (const float*)d_in[11];
  const float* rsin = (const float*)d_in[12];
  const int* seq_lens = (const int*)d_in[13];
  float* out = (float*)d_out;

  char* ws = (char*)d_ws;
  size_t off = 0;
  auto alloc = [&](size_t bytes) -> void* {
    void* p = ws + off;
    off = (off + bytes + 255) & ~(size_t)255;
    return p;
  };
  const size_t SD = (size_t)S_LEN * DIMSZ;
  const size_t WW = (size_t)DIMSZ * DIMSZ;
  bf16_t* xbf  = (bf16_t*)alloc(SD * 2);
  bf16_t* wqbf = (bf16_t*)alloc(WW * 2);
  bf16_t* wkbf = (bf16_t*)alloc(WW * 2);
  bf16_t* wvbf = (bf16_t*)alloc(WW * 2);
  bf16_t* wobf = (bf16_t*)alloc(WW * 2);
  float*  qraw = (float*)alloc(SD * 4);
  float*  kraw = (float*)alloc(SD * 4);
  float*  vraw = (float*)alloc(SD * 4);
  bf16_t* qh   = (bf16_t*)alloc(SD * 2);
  bf16_t* kh   = (bf16_t*)alloc(SD * 2);
  bf16_t* vt   = (bf16_t*)alloc(SD * 2);
  bf16_t* abuf = (bf16_t*)alloc(SD * 2);

  // 1) bf16 conversions
  cvt_f32_bf16<<<(int)((SD + 255) / 256), 256, 0, stream>>>(x, xbf, (int)SD);
  cvt_f32_bf16<<<(int)((WW + 255) / 256), 256, 0, stream>>>(Wq, wqbf, (int)WW);
  cvt_f32_bf16<<<(int)((WW + 255) / 256), 256, 0, stream>>>(Wk, wkbf, (int)WW);
  cvt_f32_bf16<<<(int)((WW + 255) / 256), 256, 0, stream>>>(Wv, wvbf, (int)WW);
  cvt_f32_bf16<<<(int)((WW + 255) / 256), 256, 0, stream>>>(Wo, wobf, (int)WW);

  // 2) QKV projections (WMMA GEMM, f32 out with bias)
  dim3 gg(DIMSZ / 256, S_LEN / 128);
  gemm_bf16_nt<<<gg, 256, 0, stream>>>(xbf, wqbf, bq, qraw, S_LEN, DIMSZ, DIMSZ);
  gemm_bf16_nt<<<gg, 256, 0, stream>>>(xbf, wkbf, bk, kraw, S_LEN, DIMSZ, DIMSZ);
  gemm_bf16_nt<<<gg, 256, 0, stream>>>(xbf, wvbf, bv, vraw, S_LEN, DIMSZ, DIMSZ);

  // 3) RMSNorm + RoPE + pack (Q, K); transpose-pack V
  norm_rope_pack<<<S_LEN, 256, 0, stream>>>(qraw, nqw, rcos, rsin, qh);
  norm_rope_pack<<<S_LEN, 256, 0, stream>>>(kraw, nkw, rcos, rsin, kh);
  pack_v<<<S_LEN, 256, 0, stream>>>(vraw, vt);

  // 4) flash attention
  attn_fwd<<<dim3(S_LEN / 64, NHEAD), 128, 0, stream>>>(qh, kh, vt, abuf, seq_lens);

  // 5) output projection -> f32 d_out
  gemm_bf16_nt<<<gg, 256, 0, stream>>>(abuf, wobf, bo, out, S_LEN, DIMSZ, DIMSZ);

  (void)in_sizes; (void)n_in; (void)out_size; (void)ws_size;
}
